// SDREngramMemory_41257455846035
// MI455X (gfx1250) — matrix operand order repaired and checked by reading, not compiled, not run on hardware
//
#include <hip/hip_runtime.h>

#define SDR_SIZE    2048
#define CAPACITY    131072
#define N_ACTIVE    40
#define CONTENT_DIM 384
#define THRESHOLD   0.3f
#define TOP_K       32

typedef __attribute__((ext_vector_type(16))) _Float16 v16h;
typedef __attribute__((ext_vector_type(8)))  float    v8f;
typedef __attribute__((ext_vector_type(4)))  float    v4f;

// Load 8 contiguous fp32 (two b128 loads), convert to 8 exact fp16 halves.
// NT variant for the 1 GiB single-use sdr_bank stream (keep L2 for sim/query/content).
__device__ __forceinline__ void load8_cvt_nt(const float* __restrict__ p, v16h& d, const int base) {
    const v4f f0 = __builtin_nontemporal_load(reinterpret_cast<const v4f*>(p));
    const v4f f1 = __builtin_nontemporal_load(reinterpret_cast<const v4f*>(p + 4));
    d[base + 0] = (_Float16)f0[0]; d[base + 1] = (_Float16)f0[1];
    d[base + 2] = (_Float16)f0[2]; d[base + 3] = (_Float16)f0[3];
    d[base + 4] = (_Float16)f1[0]; d[base + 5] = (_Float16)f1[1];
    d[base + 6] = (_Float16)f1[2]; d[base + 7] = (_Float16)f1[3];
}

__device__ __forceinline__ void load8_cvt(const float* __restrict__ p, v16h& d, const int base) {
    const v4f f0 = *reinterpret_cast<const v4f*>(p);
    const v4f f1 = *reinterpret_cast<const v4f*>(p + 4);
    d[base + 0] = (_Float16)f0[0]; d[base + 1] = (_Float16)f0[1];
    d[base + 2] = (_Float16)f0[2]; d[base + 3] = (_Float16)f0[3];
    d[base + 4] = (_Float16)f1[0]; d[base + 5] = (_Float16)f1[1];
    d[base + 6] = (_Float16)f1[2]; d[base + 7] = (_Float16)f1[3];
}

// Each wave computes overlap for a 16-row tile of sdr_bank via v_wmma_f32_16x16x32_f16.
// A = 16x32 bank tile, B = query chunk replicated into all 16 columns so every
// D column equals the overlap vector. 0/1 data is exact in f16; overlap <= 40 is
// exact in f32 accumulation.
__global__ __launch_bounds__(256) void overlap_kernel(
        const float* __restrict__ query,
        const float* __restrict__ bank,
        const unsigned char* __restrict__ valid,
        float* __restrict__ sim) {
    const int lane = threadIdx.x & 31;
    const int wave = threadIdx.x >> 5;
    const int tile = blockIdx.x * 8 + wave;   // 16 rows per tile
    const int row0 = tile * 16;
    const int hi   = (lane >= 16) ? 1 : 0;    // K-half this lane supplies
    const int m    = lane & 15;               // matrix row within tile
    const float* rowp = bank + (size_t)(row0 + m) * SDR_SIZE;

    v8f c = {};
    #pragma unroll 2
    for (int k0 = 0; k0 < SDR_SIZE; k0 += 32) {
        v16h a, b;
        // A 16-bit 16x32 layout: lanes 0-15 -> K = k0+[0..7], k0+[16..23];
        //                        lanes16-31 -> K = k0+[8..15], k0+[24..31]
        const int ka = k0 + hi * 8;
        load8_cvt_nt(rowp + ka,      a, 0);
        load8_cvt_nt(rowp + ka + 16, a, 8);
        // B 16-bit 32x16 layout (same K order as A): lanes 0-15 -> K = k0+[0..7], k0+[16..23];
        //                                           lanes16-31 -> K = k0+[8..15], k0+[24..31]
        const int kb = k0 + hi * 8;
        load8_cvt(query + kb,      b, 0);
        load8_cvt(query + kb + 16, b, 8);
        c = __builtin_amdgcn_wmma_f32_16x16x32_f16(
                /*neg_a=*/false, a, /*neg_b=*/false, b,
                /*c_mod=*/(short)0, c, /*reuse_a=*/false, /*reuse_b=*/false);
    }
    // All D columns identical: lane L holds overlap[row0 + r + 8*(L>=16)] in c[r].
    if (m == 0) {
        const int base = row0 + 8 * hi;
        #pragma unroll
        for (int r = 0; r < 8; ++r) {
            float s = c[r] / (float)N_ACTIVE;   // true division: bit-match reference
            s = valid[base + r] ? s : -__builtin_inff();
            sim[base + r] = s;
        }
    }
}

__device__ __forceinline__ unsigned long long wave_max_u64(unsigned long long v) {
    #pragma unroll
    for (int off = 16; off > 0; off >>= 1) {
        unsigned lo = (unsigned)__shfl_xor((int)(v & 0xFFFFFFFFu), off, 32);
        unsigned hv = (unsigned)__shfl_xor((int)(v >> 32),         off, 32);
        unsigned long long o = ((unsigned long long)hv << 32) | lo;
        if (o > v) v = o;
    }
    return v;
}

// Exact top-32 with jax.lax.top_k ordering: value desc, ties -> lower index.
// Single 1024-thread block; 32 argmax passes over the L2-resident sim array.
__global__ __launch_bounds__(1024) void topk32_kernel(
        float* __restrict__ sim,
        float* __restrict__ out_sim,
        int*   __restrict__ idx_ws,
        float* __restrict__ keep_ws) {
    __shared__ unsigned long long red[32];
    __shared__ int   sel_idx[TOP_K];
    __shared__ float sel_val[TOP_K];
    const int tid  = threadIdx.x;
    const int lane = tid & 31;
    const int wid  = tid >> 5;

    for (int pass = 0; pass < TOP_K; ++pass) {
        unsigned long long best = 0ull;
        for (int i = tid; i < CAPACITY; i += 1024) {
            unsigned fb = __float_as_uint(sim[i]);
            // monotonic total order for floats as unsigned
            fb = (fb & 0x80000000u) ? ~fb : (fb | 0x80000000u);
            unsigned long long key =
                ((unsigned long long)fb << 32) | (unsigned)(~i);
            if (key > best) best = key;
        }
        best = wave_max_u64(best);
        if (lane == 0) red[wid] = best;
        __syncthreads();
        if (wid == 0) {
            best = wave_max_u64(red[lane]);
            if (lane == 0) {
                int idx = (int)(~(unsigned)(best & 0xFFFFFFFFu));
                sel_idx[pass] = idx;
                sel_val[pass] = sim[idx];
                sim[idx] = -__builtin_inff();   // mask winner for next pass
            }
        }
        __syncthreads();   // publishes sim[idx] update + shared state to block
    }

    if (tid < TOP_K) {
        float v = sel_val[tid];
        bool keep = (v >= THRESHOLD);
        out_sim[tid] = keep ? v : 0.0f;
        idx_ws[tid]  = sel_idx[tid];
        keep_ws[tid] = keep ? 1.0f : 0.0f;
    }
}

__global__ __launch_bounds__(128) void gather_kernel(
        const float* __restrict__ content,
        const int*   __restrict__ idx_ws,
        const float* __restrict__ keep_ws,
        float* __restrict__ out) {
    const int r   = blockIdx.x;
    const int idx = idx_ws[r];
    const float s = keep_ws[r];
    const float* src = content + (size_t)idx * CONTENT_DIM;
    for (int col = threadIdx.x; col < CONTENT_DIM; col += 128)
        out[r * CONTENT_DIM + col] = src[col] * s;
}

extern "C" void kernel_launch(void* const* d_in, const int* in_sizes, int n_in,
                              void* d_out, int out_size, void* d_ws, size_t ws_size,
                              hipStream_t stream) {
    const float*         query   = (const float*)d_in[0];         // [2048]
    const float*         bank    = (const float*)d_in[1];         // [131072, 2048]
    const float*         content = (const float*)d_in[2];         // [131072, 384]
    const unsigned char* valid   = (const unsigned char*)d_in[3]; // [131072] bool
    // d_in[4] = top_k scalar on device (unreadable during capture); fixed at 32.

    float* out = (float*)d_out;   // [32*384 contents] + [32 top_sim]

    float* sim     = (float*)d_ws;                // 131072 floats
    int*   idx_ws  = (int*)(sim + CAPACITY);      // 32 ints
    float* keep_ws = (float*)(idx_ws + 64);       // 32 floats (padded)

    // 1 GiB NT stream of sdr_bank through WMMA: 8192 tiles of 16 rows, 8 waves/block.
    overlap_kernel<<<CAPACITY / 128, 256, 0, stream>>>(query, bank, valid, sim);

    // Exact ordered top-32 (+ threshold) -> tail of d_out and gather metadata.
    topk32_kernel<<<1, 1024, 0, stream>>>(sim, out + TOP_K * CONTENT_DIM,
                                          idx_ws, keep_ws);

    // Gather/zero the 32 content rows.
    gather_kernel<<<TOP_K, 128, 0, stream>>>(content, idx_ws, keep_ws, out);
}
